// Quantize_19765439496196
// MI455X (gfx1250) — matrix operand order repaired and checked by reading, not compiled, not run on hardware
//
#include <hip/hip_runtime.h>

#define DIM    256
#define NEMB   8192
#define NROWS  16384               // 16*1024
#define NUMEL  4194304             // NROWS*DIM
#define KTILES 512                 // NEMB/16
#define BETA_F 0.25f
#define AROW   1040                // staged codebook row pitch in bytes (512 halves*2 + 16 pad)

typedef __attribute__((ext_vector_type(16))) __bf16       v16bf;
typedef __attribute__((ext_vector_type(8)))  float        v8f;
typedef __attribute__((ext_vector_type(4)))  unsigned int u32x4;

union V16 { v16bf v; unsigned short s[16]; u32x4 q[2]; };

__device__ __forceinline__ unsigned short f2bf(float f) {
    unsigned u = __float_as_uint(f);
    return (unsigned short)((u + 0x7FFFu + ((u >> 16) & 1u)) >> 16);   // RNE
}
__device__ __forceinline__ float bf2f(unsigned short h) {
    return __uint_as_float(((unsigned)h) << 16);
}

__device__ __forceinline__ v8f wmma_bf16(v16bf a, v16bf b, v8f c) {
    return __builtin_amdgcn_wmma_f32_16x16x32_bf16(false, a, false, b, (short)0, c, false, false);
}

__device__ __forceinline__ v16bf ld2(const unsigned char* p0, const unsigned char* p1) {
    V16 u;
    u.q[0] = *(const u32x4*)p0;
    u.q[1] = *(const u32x4*)p1;
    return u.v;
}

// ---------------------------------------------------------------------------
// K1: codebook[k,j] = sum_d embed[k,d]*proj_w[j,d] + b[j], split-bf16 WMMA.
// Writes f32 codebook and hi/lo bf16 augmented codebook C2 [NEMB x 512].
// grid 1024 x 256 (8 waves/block, 1 16x16 tile per wave, 8192 tiles total)
// ---------------------------------------------------------------------------
__global__ __launch_bounds__(256) void k1_project(
    const float* __restrict__ embed, const float* __restrict__ projw,
    const float* __restrict__ projb, float* __restrict__ cb,
    unsigned short* __restrict__ C2) {
    const int tid  = threadIdx.x;
    const int lane = tid & 31, wave = tid >> 5;
    const int r16  = lane & 15, half = lane >> 4;
    const int tile = blockIdx.x * 8 + wave;         // 0..8191
    const int kt   = tile >> 4;                     // code tile 0..511
    const int jt   = tile & 15;                     // output-dim tile 0..15
    const int mrow = kt * 16 + r16;                 // A row (code)
    const int jcol = jt * 16 + r16;                 // B column (output dim)
    const float* er = embed + (size_t)mrow * DIM;
    const float* wr = projw + (size_t)jcol * DIM;
    const int aoff = half * 8;                      // A lane K base
    const int boff = half * 16;                     // unused; B also uses aoff as its lane K base

    v8f acc = {0, 0, 0, 0, 0, 0, 0, 0};
#pragma unroll
    for (int j = 0; j < 8; ++j) {
        V16 ah, al, bh, bl;
#pragma unroll
        for (int e = 0; e < 16; ++e) {
            const int ka = j * 32 + aoff + (e & 7) + ((e >> 3) << 4);
            const float va = er[ka];
            const unsigned short ha = f2bf(va);
            ah.s[e] = ha;
            al.s[e] = f2bf(va - bf2f(ha));
            const float vb = wr[j * 32 + aoff + (e & 7) + ((e >> 3) << 4)];
            const unsigned short hb = f2bf(vb);
            bh.s[e] = hb;
            bl.s[e] = f2bf(vb - bf2f(hb));
        }
        acc = wmma_bf16(ah.v, bh.v, acc);   // hi*hi
        acc = wmma_bf16(al.v, bh.v, acc);   // lo*hi
        acc = wmma_bf16(ah.v, bl.v, acc);   // hi*lo
    }
    const float bias = projb[jcol];
#pragma unroll
    for (int r = 0; r < 8; ++r) {
        const int m = kt * 16 + (half << 3) + r;    // D VGPR r -> M = r + 8*half
        const float c = acc[r] + bias;
        cb[(size_t)m * DIM + jcol] = c;
        const unsigned short h = f2bf(c);
        C2[(size_t)m * 512 + jcol]       = h;                   // hi block
        C2[(size_t)m * 512 + 256 + jcol] = f2bf(c - bf2f(h));   // lo block
    }
}

// ---------------------------------------------------------------------------
// K2: cnorm[k] = 0.5 * ||codebook[k]||^2.   grid NEMB x 256
// ---------------------------------------------------------------------------
__global__ void k2_norms(const float* __restrict__ cb, float* __restrict__ cnorm) {
    __shared__ float red[256];
    const int k = blockIdx.x, tid = threadIdx.x;
    const float v = cb[(size_t)k * DIM + tid];
    red[tid] = v * v;
    __syncthreads();
    for (int s = 128; s > 0; s >>= 1) {
        if (tid < s) red[tid] += red[tid + s];
        __syncthreads();
    }
    if (tid == 0) cnorm[k] = 0.5f * red[0];
}

// ---------------------------------------------------------------------------
// K3: main GEMM + argmax.  Workgroup = 8 waves, wave w owns x row-tile
// (blockIdx*8+w); x fragments live in registers; codebook tiles staged in
// double-buffered LDS shared by all 8 waves.   grid 128 x 256
// ---------------------------------------------------------------------------
__global__ __launch_bounds__(256) void k3_argmin(
    const float* __restrict__ x, const unsigned short* __restrict__ C2,
    const float* __restrict__ cnorm, int* __restrict__ widx,
    float* __restrict__ outIdx) {
    __shared__ __align__(16) unsigned char sA[2][16 * AROW];
    __shared__ float sN[2][16];

    const int tid  = threadIdx.x;
    const int lane = tid & 31, wave = tid >> 5;
    const int r16  = lane & 15, half = lane >> 4;
    const int n0   = (blockIdx.x * 8 + wave) * 16;

    // Build B (x) fragments: 8 hi + 8 lo v16bf, register-resident (128 VGPRs).
    const float* xr = x + (size_t)(n0 + r16) * DIM;
    v16bf xhi[8], xlo[8];
#pragma unroll
    for (int j = 0; j < 8; ++j) {
        V16 uh, ul;
        const int kb = j * 32 + half * 8;           // B lane pattern: 8 at kb, 8 at kb+16
#pragma unroll
        for (int e = 0; e < 16; ++e) {
            const float v = xr[kb + (e & 7) + ((e >> 3) << 4)];
            const unsigned short h = f2bf(v);
            uh.s[e] = h;
            ul.s[e] = f2bf(v - bf2f(h));
        }
        xhi[j] = uh.v;
        xlo[j] = ul.v;
    }

    auto stage = [&](int kt, int buf) {
#pragma unroll
        for (int i = 0; i < 4; ++i) {
            const int c = tid + i * 256;            // 1024 chunks of 16B
            const int row = c >> 6, cho = c & 63;
            const u32x4* g = (const u32x4*)(C2 + (size_t)(kt * 16 + row) * 512) + cho;
            *(u32x4*)(&sA[buf][row * AROW + cho * 16]) = *g;
        }
        if (tid < 16) sN[buf][tid] = cnorm[kt * 16 + tid];
    };

    float bs = -3.4e38f;
    int   bi = 0;

    stage(0, 0);
    __syncthreads();
    for (int kt = 0; kt < KTILES; ++kt) {
        if (kt + 1 < KTILES) stage(kt + 1, (kt + 1) & 1);
        if (kt + 2 < KTILES)
            __builtin_prefetch((const void*)((const unsigned char*)(C2 + (size_t)(kt + 2) * 16 * 512) + tid * 64), 0, 1);

        const unsigned char* rowp = &sA[kt & 1][r16 * AROW];
        v8f acc = {0, 0, 0, 0, 0, 0, 0, 0};
#pragma unroll
        for (int j = 0; j < 8; ++j) {
            const int kb = (j * 32 + half * 8) * 2;               // byte offset in hi block
            const v16bf ah = ld2(rowp + kb,       rowp + kb + 32);
            const v16bf al = ld2(rowp + 512 + kb, rowp + 512 + kb + 32);
            acc = wmma_bf16(ah, xhi[j], acc);
            acc = wmma_bf16(al, xhi[j], acc);
            acc = wmma_bf16(ah, xlo[j], acc);
        }
#pragma unroll
        for (int r = 0; r < 8; ++r) {
            const float sc = acc[r] - sN[kt & 1][(half << 3) + r];
            if (sc > bs) { bs = sc; bi = kt * 16 + (half << 3) + r; }
        }
        __syncthreads();
    }

    // combine the two lane halves (codes m..m+7 vs m+8..m+15 of same x row)
    const float obs = __shfl_xor(bs, 16, 32);
    const int   obi = __shfl_xor(bi, 16, 32);
    if (obs > bs || (obs == bs && obi < bi)) { bs = obs; bi = obi; }
    if (lane < 16) {
        widx[n0 + lane]   = bi;
        outIdx[n0 + lane] = (float)bi;
    }
}

// ---------------------------------------------------------------------------
// K4: gather z_quantize + per-block squared-error partials.  grid 256 x 256
// ---------------------------------------------------------------------------
__global__ void k4_gather(const float* __restrict__ x, const float* __restrict__ cb,
                          const int* __restrict__ widx, float* __restrict__ zq,
                          float* __restrict__ partials) {
    __shared__ float red[256];
    const int tid = threadIdx.x;
    const int r0  = blockIdx.x * 64;
    float s = 0.f;
    for (int r = r0; r < r0 + 64; ++r) {
        const int idx = widx[r];
        const float z  = cb[(size_t)idx * DIM + tid];
        const float xv = x[(size_t)r * DIM + tid];
        zq[(size_t)r * DIM + tid] = z;
        const float d = z - xv;
        s += d * d;
    }
    red[tid] = s;
    __syncthreads();
    for (int t = 128; t > 0; t >>= 1) {
        if (tid < t) red[tid] += red[tid + t];
        __syncthreads();
    }
    if (tid == 0) partials[blockIdx.x] = red[0];
}

// ---------------------------------------------------------------------------
// K5: diff = (1+beta) * mean((zq - x)^2).  grid 1 x 256
// ---------------------------------------------------------------------------
__global__ void k5_final(const float* __restrict__ partials, float* __restrict__ diff) {
    __shared__ float red[256];
    const int tid = threadIdx.x;
    red[tid] = partials[tid];
    __syncthreads();
    for (int t = 128; t > 0; t >>= 1) {
        if (tid < t) red[tid] += red[tid + t];
        __syncthreads();
    }
    if (tid == 0) *diff = red[0] * (1.0f + BETA_F) / (float)NUMEL;
}

extern "C" void kernel_launch(void* const* d_in, const int* in_sizes, int n_in,
                              void* d_out, int out_size, void* d_ws, size_t ws_size,
                              hipStream_t stream) {
    const float* x     = (const float*)d_in[0];   // [16,1024,256]
    const float* embed = (const float*)d_in[1];   // [8192,256]
    const float* projw = (const float*)d_in[2];   // [256,256]
    const float* projb = (const float*)d_in[3];   // [256]

    float* out    = (float*)d_out;
    float* zq     = out;                 // NUMEL
    float* diff   = out + NUMEL;         // 1
    float* outIdx = out + NUMEL + 1;     // NROWS (indices as exact floats)

    char* ws = (char*)d_ws;
    float*          cb       = (float*)ws;                                    // 8 MB
    unsigned short* C2       = (unsigned short*)(ws + (8u << 20));            // 8 MB
    float*          cnorm    = (float*)(ws + (16u << 20));                    // 32 KB
    int*            widx     = (int*)  (ws + (16u << 20) + 32768);            // 64 KB
    float*          partials = (float*)(ws + (16u << 20) + 32768 + 65536);    // 1 KB

    k1_project<<<1024, 256, 0, stream>>>(embed, projw, projb, cb, C2);
    k2_norms  <<<NEMB, 256, 0, stream>>>(cb, cnorm);
    k3_argmin <<<128,  256, 0, stream>>>(x, C2, cnorm, widx, outIdx);
    k4_gather <<<256,  256, 0, stream>>>(x, cb, widx, zq, partials);
    k5_final  <<<1,    256, 0, stream>>>(partials, diff);
}
